// ResNet34Mvt_36240934043821
// MI455X (gfx1250) — matrix operand order repaired and checked by reading, not compiled, not run on hardware
//
#include <hip/hip_runtime.h>
#include <hip/hip_bf16.h>
#include <math.h>

// ---------------------------------------------------------------------------
// Types for CDNA5 WMMA / TDM
// ---------------------------------------------------------------------------
typedef __attribute__((ext_vector_type(16))) __bf16 v16bf;
typedef __attribute__((ext_vector_type(2)))  __bf16 v2bf;
typedef __attribute__((ext_vector_type(8)))  float  v8f;
typedef __attribute__((ext_vector_type(4)))  unsigned v4u;
typedef __attribute__((ext_vector_type(8)))  int v8i;
typedef __attribute__((ext_vector_type(4)))  int v4i;

// ---------------------------------------------------------------------------
// Device helpers
// ---------------------------------------------------------------------------
__device__ __forceinline__ float gelu_exact(float x) {
  return 0.5f * x * (1.0f + erff(x * 0.70710678118654752f));
}

// Tensor Data Mover: async up-to-3D tile load (global -> LDS), bf16 elements.
// Descriptor layout per CDNA5 ISA (group0 128b, group1 256b).
// 6-arg builtin form (amdgpu-toolchain clang-23).
//   tile dims (t0,t1,t2) in elements; tensor dims (td0,td1) for OOB zero-fill;
//   s0 = stride between dim-1 lines, s1 = stride between dim-2 planes (elements).
// LDS receives the tile contiguously in x-then-y-then-z order.
__device__ __forceinline__ void tdm_load(unsigned lds_off, const void* gptr,
                                         unsigned t0, unsigned t1, unsigned t2,
                                         unsigned td0, unsigned td1,
                                         unsigned long long s0,
                                         unsigned long long s1) {
  unsigned long long ga = (unsigned long long)(uintptr_t)gptr;
  v4u g0;
  g0[0] = 1u;                                       // count=1, user mode
  g0[1] = lds_off;                                  // lds_addr
  g0[2] = (unsigned)ga;                             // global_addr[31:0]
  g0[3] = (unsigned)((ga >> 32) & 0x01FFFFFFu) | (2u << 30);  // addr[56:32], type=2
  v8i g1;
  g1[0] = (int)(1u << 16);                          // wg_mask=0, data_size=1 (2B)
  g1[1] = (int)((td0 & 0xFFFFu) << 16);             // tensor_dim0[15:0]
  g1[2] = (int)(((td0 >> 16) & 0xFFFFu) | ((td1 & 0xFFFFu) << 16));
  g1[3] = (int)(((td1 >> 16) & 0xFFFFu) | ((t0 & 0xFFFFu) << 16));   // tile_dim0
  g1[4] = (int)((t1 & 0xFFFFu) | ((t2 & 0xFFFFu) << 16));            // tile_dim1/2
  g1[5] = (int)(unsigned)(s0 & 0xFFFFFFFFull);      // tensor_dim0_stride[31:0]
  g1[6] = (int)(unsigned)(((s0 >> 32) & 0xFFFFull) | ((s1 & 0xFFFFull) << 16));
  g1[7] = (int)(unsigned)((s1 >> 16) & 0xFFFFFFFFull);
  v4i z4 = {0, 0, 0, 0};
  v8i z8 = {0, 0, 0, 0, 0, 0, 0, 0};
  __builtin_amdgcn_tensor_load_to_lds(g0, g1, z4, z4, z8, 0);
}

// ---------------------------------------------------------------------------
// Core GEMM:  C[M,N] = A[M,K](bf16) * B[K,N](bf16)  (+bias, +gelu, +residual)
// 64x64 tile / block, 4 waves, each wave 2x2 WMMA 16x16x32 bf16->f32 tiles.
// Double-buffered TDM pipeline: wave 0 issues the next K-step's A/B tile DMAs
// into the ping-pong buffer right after the barrier, overlapping TDM with WMMA.
// B tile is DMA'd column-major (3D descriptor: 1 x 32(k) x 64(n)) so both A and
// B fragments are contiguous 16-byte LDS reads (ds_load_b128).
// ---------------------------------------------------------------------------
__launch_bounds__(128)
__global__ void k_gemm_bf16(const __bf16* __restrict__ A, const __bf16* __restrict__ Bm,
                            const float* __restrict__ bias, const float* __restrict__ resid,
                            float* __restrict__ Cf, __bf16* __restrict__ Cb,
                            int M, int N, int K, int lda, int ldb, int ldc, int dogelu) {
  __shared__ __bf16 sA[2][64 * 32];    // [m][k]
  __shared__ __bf16 sBt[2][64 * 32];   // [n][k]  (column-major tile)
  const int tid  = threadIdx.x;
  const int wid  = tid >> 5;
  const int lane = tid & 31;
  const int m0 = blockIdx.y * 64;
  const int n0 = blockIdx.x * 64;
  const int wm = (wid & 1) * 32;
  const int wn = (wid >> 1) * 32;
  const int l15 = lane & 15;
  const int g   = lane >> 4;

  auto load_tiles = [&](int buf, int k0) {
    // A: 64 rows x 32 k, row-major in LDS
    tdm_load((unsigned)(uintptr_t)&sA[buf][0], A + (size_t)m0 * lda + k0,
             32u, 64u, 0u, (unsigned)(K - k0), (unsigned)(M - m0),
             (unsigned long long)lda, 0ull);
    // B: column-major tile; x=1 elem, y=32 over k (stride ldb), z=64 over n (stride 1)
    tdm_load((unsigned)(uintptr_t)&sBt[buf][0], Bm + (size_t)k0 * ldb + n0,
             1u, 32u, 64u, (unsigned)(K - k0), (unsigned)(N - n0),
             (unsigned long long)ldb, 1ull);
  };

  v8f acc[2][2] = {};
  if (wid == 0) load_tiles(0, 0);

  int cur = 0;
  for (int k0 = 0; k0 < K; k0 += 32, cur ^= 1) {
    if (wid == 0) __builtin_amdgcn_s_wait_tensorcnt(0);
    __syncthreads();   // current tiles visible; previous buffer free to overwrite
    if (wid == 0 && (k0 + 32) < K) load_tiles(cur ^ 1, k0 + 32);

    v16bf afrag[2], bfrag[2];
#pragma unroll
    for (int i = 0; i < 2; ++i) {
      // A fragment: lane l<16 -> row M=l; VGPR v holds K pairs {0..7,16..23}(+8 hi)
      const __bf16* ar = &sA[cur][(size_t)(wm + i * 16 + l15) * 32];
#pragma unroll
      for (int v = 0; v < 8; ++v) {
        int kk = (v < 4) ? (2 * v + 8 * g) : (16 + 2 * (v - 4) + 8 * g);
        v2bf p = *(const v2bf*)(ar + kk);
        afrag[i][2 * v]     = p.x;
        afrag[i][2 * v + 1] = p.y;
      }
      // B fragment: lane -> column N=l15; VGPR v holds K pairs {0..7,16..23}(+8 hi),
      // matching the A fragment K order
      const __bf16* br = &sBt[cur][(size_t)(wn + i * 16 + l15) * 32];
#pragma unroll
      for (int v = 0; v < 8; ++v) {
        int kk = (v < 4) ? (2 * v + 8 * g) : (16 + 2 * (v - 4) + 8 * g);
        v2bf p = *(const v2bf*)(br + kk);
        bfrag[i][2 * v]     = p.x;
        bfrag[i][2 * v + 1] = p.y;
      }
    }
#pragma unroll
    for (int i = 0; i < 2; ++i)
#pragma unroll
      for (int j = 0; j < 2; ++j)
        acc[i][j] = __builtin_amdgcn_wmma_f32_16x16x32_bf16(
            false, afrag[i], false, bfrag[j], (short)0, acc[i][j], false, false);
  }

  // Epilogue: C/D layout -> lane l<16: col=l, rows r(+8 for hi lane half)
#pragma unroll
  for (int i = 0; i < 2; ++i)
#pragma unroll
    for (int j = 0; j < 2; ++j) {
      int col = n0 + wn + j * 16 + l15;
      if (col >= N) continue;
      float bv = bias ? bias[col] : 0.0f;
#pragma unroll
      for (int r = 0; r < 8; ++r) {
        int row = m0 + wm + i * 16 + 8 * g + r;
        if (row >= M) continue;
        float v = acc[i][j][r] + bv;
        if (dogelu) v = gelu_exact(v);
        size_t o = (size_t)row * ldc + col;
        if (resid) v += resid[o];
        if (Cf) Cf[o] = v;
        if (Cb) Cb[o] = (__bf16)v;
      }
    }
}

// ---------------------------------------------------------------------------
// Elementwise / small kernels (VALU; negligible FLOPs)
// ---------------------------------------------------------------------------
__global__ void k_cvt_bf16(const float* in, __bf16* out, long n) {
  long i = (long)blockIdx.x * blockDim.x + threadIdx.x;
  if (i < n) out[i] = (__bf16)in[i];
}

// out[r*cols+c] = in[r*si + c*sj]  (optional f32 / bf16 outputs)
__global__ void k_gather2d(const float* in, float* outf, __bf16* outb,
                           long rows, long cols, long si, long sj) {
  long i = (long)blockIdx.x * blockDim.x + threadIdx.x;
  if (i >= rows * cols) return;
  long r = i / cols, c = i % cols;
  float v = in[r * si + c * sj];
  if (outf) outf[i] = v;
  if (outb) outb[i] = (__bf16)v;
}

__global__ void k_layernorm(const float* x, const float* g, const float* b,
                            float* outf, __bf16* outb, long rows, int C) {
  long r = (long)blockIdx.x * blockDim.x + threadIdx.x;
  if (r >= rows) return;
  const float* xr = x + r * (long)C;
  float m = 0.f;
  for (int c = 0; c < C; ++c) m += xr[c];
  m /= (float)C;
  float v = 0.f;
  for (int c = 0; c < C; ++c) { float d = xr[c] - m; v += d * d; }
  v /= (float)C;
  float inv = rsqrtf(v + 1e-5f);
  for (int c = 0; c < C; ++c) {
    float y = (xr[c] - m) * inv * g[c] + b[c];
    if (outf) outf[r * (long)C + c] = y;
    if (outb) outb[r * (long)C + c] = (__bf16)y;
  }
}

__global__ void k_softmax(const float* S, __bf16* P, long rows, int cols, float scale) {
  long r = (long)blockIdx.x * blockDim.x + threadIdx.x;
  if (r >= rows) return;
  const float* sr = S + r * (long)cols;
  __bf16* pr = P + r * (long)cols;
  float mx = -3.0e38f;
  for (int c = 0; c < cols; ++c) mx = fmaxf(mx, sr[c] * scale);
  float sum = 0.f;
  for (int c = 0; c < cols; ++c) sum += expf(sr[c] * scale - mx);
  float inv = 1.0f / sum;
  for (int c = 0; c < cols; ++c) pr[c] = (__bf16)(expf(sr[c] * scale - mx) * inv);
}

__global__ void k_conv2d(const float* in, const float* w, const float* bias,
                         const float* resid, float* out,
                         int B, int Cin, int H, int W, int Cout, int OH, int OW,
                         int KH, int KW, int stride, int pad, int dil, int groups,
                         int in_coff, int in_Ctot) {
  long idx = (long)blockIdx.x * blockDim.x + threadIdx.x;
  long tot = (long)B * Cout * OH * OW;
  if (idx >= tot) return;
  int ow = (int)(idx % OW);
  long t = idx / OW;
  int oh = (int)(t % OH); t /= OH;
  int oc = (int)(t % Cout);
  int b  = (int)(t / Cout);
  int icpg = Cin / groups, ocpg = Cout / groups;
  int gi = oc / ocpg;
  float acc = bias ? bias[oc] : 0.f;
  for (int ic = 0; ic < icpg; ++ic) {
    int icg = gi * icpg + ic;
    const float* ip = in + ((long)b * in_Ctot + in_coff + icg) * (long)H * W;
    const float* wp = w + ((long)oc * icpg + ic) * (long)KH * KW;
    for (int kh = 0; kh < KH; ++kh) {
      int ih = oh * stride - pad + kh * dil;
      if (ih < 0 || ih >= H) continue;
      for (int kw = 0; kw < KW; ++kw) {
        int iw = ow * stride - pad + kw * dil;
        if (iw < 0 || iw >= W) continue;
        acc += ip[(long)ih * W + iw] * wp[kh * KW + kw];
      }
    }
  }
  if (resid) acc += resid[idx];
  out[idx] = acc;
}

// (B,C,Npix) image -> (B,Npix,Ctot) tokens at channel offset coff
__global__ void k_img_to_tok(const float* img, float* out, int B, int Cc, int Np,
                             int Ctot, int coff) {
  long idx = (long)blockIdx.x * blockDim.x + threadIdx.x;
  if (idx >= (long)B * Cc * Np) return;
  int n = (int)(idx % Np);
  long t = idx / Np;
  int c = (int)(t % Cc);
  int b = (int)(t / Cc);
  out[((long)b * Np + n) * Ctot + coff + c] = img[idx];
}

// (B,Npix,C) tokens -> (B,C,Npix) image (+ optional image residual)
__global__ void k_tok_to_img(const float* tok, const float* addimg, float* out,
                             int B, int Cc, int Np) {
  long idx = (long)blockIdx.x * blockDim.x + threadIdx.x;
  if (idx >= (long)B * Cc * Np) return;
  int n = (int)(idx % Np);
  long t = idx / Np;
  int c = (int)(t % Cc);
  int b = (int)(t / Cc);
  float v = tok[((long)b * Np + n) * Cc + c];
  if (addimg) v += addimg[idx];
  out[idx] = v;
}

// (B*N, C) f32 -> (B,h,N,hd) bf16
__global__ void k_head_gather(const float* q, __bf16* out, int B, int heads, int N, int hd) {
  long idx = (long)blockIdx.x * blockDim.x + threadIdx.x;
  long tot = (long)B * heads * N * hd;
  if (idx >= tot) return;
  int d = (int)(idx % hd);
  long t = idx / hd;
  int n = (int)(t % N); t /= N;
  int h = (int)(t % heads);
  int b = (int)(t / heads);
  int C = heads * hd;
  out[idx] = (__bf16)q[((long)b * N + n) * C + h * hd + d];
}

// (B,Nk,2C) f32 -> kT (B,h,hd,Nk) bf16 and v (B,h,Nk,hd) bf16
__global__ void k_kt_gather(const float* kv, __bf16* kt, __bf16* v,
                            int B, int heads, int Nk, int hd) {
  long idx = (long)blockIdx.x * blockDim.x + threadIdx.x;
  long tot = (long)B * heads * Nk * hd;
  if (idx >= tot) return;
  int d = (int)(idx % hd);
  long t = idx / hd;
  int n = (int)(t % Nk); t /= Nk;
  int h = (int)(t % heads);
  int b = (int)(t / heads);
  int C = heads * hd;
  const float* row = kv + ((long)b * Nk + n) * (2L * C);
  kt[(((long)b * heads + h) * hd + d) * Nk + n] = (__bf16)row[h * hd + d];
  v [(((long)b * heads + h) * Nk + n) * hd + d] = (__bf16)row[C + h * hd + d];
}

// avg-pool fxf from token layout (B,HW,C) -> image (B,C,OH,OW)
__global__ void k_avgpool_tok(const float* tok, float* out, int B, int C, int H, int W, int f) {
  int OH = H / f, OW = W / f;
  long idx = (long)blockIdx.x * blockDim.x + threadIdx.x;
  if (idx >= (long)B * C * OH * OW) return;
  int ow = (int)(idx % OW);
  long t = idx / OW;
  int oh = (int)(t % OH); t /= OH;
  int c = (int)(t % C);
  int b = (int)(t / C);
  float s = 0.f;
  for (int i = 0; i < f; ++i)
    for (int j = 0; j < f; ++j) {
      int ih = oh * f + i, iw = ow * f + j;
      s += tok[((long)b * H * W + (long)ih * W + iw) * C + c];
    }
  out[idx] = s / (float)(f * f);
}

__global__ void k_mean_hw(const float* in, float* out, int B, int C, int HW) {
  long idx = (long)blockIdx.x * blockDim.x + threadIdx.x;
  if (idx >= (long)B * C) return;
  const float* p = in + idx * (long)HW;
  float s = 0.f;
  for (int i = 0; i < HW; ++i) s += p[i];
  out[idx] = s / (float)HW;
}

__global__ void k_final(const float* fm, const float* xg,
                        const float* fcw, const float* fcb,
                        const float* fcmw, const float* fcmb,
                        float* out, int Bn, int C, int NC) {
  long idx = (long)blockIdx.x * blockDim.x + threadIdx.x;
  if (idx >= (long)Bn * NC) return;
  int b = (int)(idx / NC), j = (int)(idx % NC);
  float nrm = 0.f;
  for (int c = 0; c < C; ++c) { float v = fm[b * C + c]; nrm += v * v; }
  nrm = sqrtf(nrm);
  float a = 0.f, l = 0.f;
  for (int c = 0; c < C; ++c) {
    a += (fm[b * C + c] / nrm) * fcmw[c * NC + j];
    l += xg[b * C + c] * fcw[c * NC + j];
  }
  out[idx] = (l + fcb[j] + a + fcmb[j]) * 0.5f;
}

// ---------------------------------------------------------------------------
// Host-side orchestration
// ---------------------------------------------------------------------------
namespace {

struct Arena {
  char* p;
  size_t off;
  void* alloc(size_t bytes) {
    void* r = p + off;
    off += (bytes + 255) & ~(size_t)255;
    return r;
  }
};

inline dim3 gcfg(long n) { return dim3((unsigned)((n + 255) / 256)); }

void gemm(hipStream_t s, const __bf16* A, const __bf16* Bm, const float* bias,
          const float* resid, float* Cf, __bf16* Cb,
          long M, long N, long K, long lda, long ldb, long ldc, int dogelu) {
  dim3 grid((unsigned)((N + 63) / 64), (unsigned)((M + 63) / 64));
  k_gemm_bf16<<<grid, dim3(128), 0, s>>>(A, Bm, bias, resid, Cf, Cb,
                                         (int)M, (int)N, (int)K,
                                         (int)lda, (int)ldb, (int)ldc, dogelu);
}

// One PVT-style transformer block on an NCHW feature map (heads=4).
void run_block(const float* ximg, const float* const* p, int B, int C, int H, int W,
               int heads, float* fimg_out, Arena scratch, hipStream_t s) {
  const int Np = H * W, c4 = C / 4, sh = H / 2, sw = W / 2, Nk = sh * sw, hd = C / heads;
  const long BN = (long)B * Np;

  auto cvtw = [&](const float* w, long n) {
    __bf16* o = (__bf16*)scratch.alloc((size_t)n * 2);
    k_cvt_bf16<<<gcfg(n), 256, 0, s>>>(w, o, n);
    return o;
  };

  // tokens + LN1
  float* x = (float*)scratch.alloc((size_t)BN * C * 4);
  k_img_to_tok<<<gcfg((long)B * C * Np), 256, 0, s>>>(ximg, x, B, C, Np, C, 0);
  __bf16* xln = (__bf16*)scratch.alloc((size_t)BN * C * 2);
  k_layernorm<<<gcfg(BN), 256, 0, s>>>(x, p[0], p[1], nullptr, xln, BN, C);

  // weights -> bf16
  __bf16* qw  = cvtw(p[2], (long)C * C);
  __bf16* kvw = cvtw(p[3], (long)C * 2 * C);
  __bf16* pjw = cvtw(p[4], (long)C * C);
  __bf16* f1w = cvtw(p[26], (long)C * 4 * C);
  __bf16* f2w = cvtw(p[28], (long)4 * C * C);

  // q projection (flattened over batch) + head split
  float* Qf = (float*)scratch.alloc((size_t)BN * C * 4);
  gemm(s, xln, qw, nullptr, nullptr, Qf, nullptr, BN, C, C, C, C, C, 0);
  __bf16* Qh = (__bf16*)scratch.alloc((size_t)BN * C * 2);
  k_head_gather<<<gcfg(BN * C), 256, 0, s>>>(Qf, Qh, B, heads, Np, hd);

  // spatial-reduction path: 4 strided convs + depthwise pos conv, concat tokens
  struct CP { int k, pad, dil; } cps[4] = {{1, 0, 1}, {3, 1, 1}, {3, 2, 2}, {3, 3, 3}};
  float* xsr = (float*)scratch.alloc((size_t)B * Nk * C * 4);
  for (int i = 0; i < 4; ++i) {
    float* srt = (float*)scratch.alloc((size_t)B * c4 * Nk * 4);
    k_conv2d<<<gcfg((long)B * c4 * Nk), 256, 0, s>>>(
        ximg, p[6 + 2 * i], p[7 + 2 * i], nullptr, srt,
        B, c4, H, W, c4, sh, sw, cps[i].k, cps[i].k, 2, cps[i].pad, cps[i].dil, 1,
        i * c4, C);
    float* post = (float*)scratch.alloc((size_t)B * c4 * Nk * 4);
    k_conv2d<<<gcfg((long)B * c4 * Nk), 256, 0, s>>>(
        srt, p[14 + 2 * i], p[15 + 2 * i], srt, post,
        B, c4, sh, sw, c4, sh, sw, 3, 3, 1, 1, 1, c4, 0, c4);
    k_img_to_tok<<<gcfg((long)B * c4 * Nk), 256, 0, s>>>(post, xsr, B, c4, Nk, C, i * c4);
  }
  __bf16* xsrb = (__bf16*)scratch.alloc((size_t)B * Nk * C * 2);
  k_layernorm<<<gcfg((long)B * Nk), 256, 0, s>>>(xsr, p[22], p[23], nullptr, xsrb,
                                                 (long)B * Nk, C);

  // kv projection + split into kT / v (bf16)
  float* KV = (float*)scratch.alloc((size_t)B * Nk * 2 * C * 4);
  gemm(s, xsrb, kvw, nullptr, nullptr, KV, nullptr, (long)B * Nk, 2 * C, C, C, 2 * C,
       2 * C, 0);
  __bf16* KT = (__bf16*)scratch.alloc((size_t)B * C * Nk * 2);
  __bf16* V  = (__bf16*)scratch.alloc((size_t)B * C * Nk * 2);
  k_kt_gather<<<gcfg((long)B * C * Nk), 256, 0, s>>>(KV, KT, V, B, heads, Nk, hd);

  // streamed attention per (b,h): scores buffer reused
  float* Sf = (float*)scratch.alloc((size_t)Np * Nk * 4);
  __bf16* Pb = (__bf16*)scratch.alloc((size_t)Np * Nk * 2);
  __bf16* AO = (__bf16*)scratch.alloc((size_t)BN * C * 2);
  float scale = 1.0f / sqrtf((float)hd);
  for (int b = 0; b < B; ++b)
    for (int h = 0; h < heads; ++h) {
      long bh = (long)b * heads + h;
      const __bf16* q  = Qh + bh * Np * hd;
      const __bf16* kt = KT + bh * hd * Nk;
      const __bf16* vv = V + bh * Nk * hd;
      gemm(s, q, kt, nullptr, nullptr, Sf, nullptr, Np, Nk, hd, hd, Nk, Nk, 0);
      k_softmax<<<gcfg(Np), 256, 0, s>>>(Sf, Pb, Np, Nk, scale);
      gemm(s, Pb, vv, nullptr, nullptr, nullptr, AO + ((long)b * Np * C + h * hd),
           Np, hd, Nk, Nk, hd, C, 0);
    }

  // output proj + residual; LN2; MLP with GELU; residual
  float* Xa = (float*)scratch.alloc((size_t)BN * C * 4);
  gemm(s, AO, pjw, p[5], x, Xa, nullptr, BN, C, C, C, C, C, 0);
  __bf16* xln2 = (__bf16*)scratch.alloc((size_t)BN * C * 2);
  k_layernorm<<<gcfg(BN), 256, 0, s>>>(Xa, p[24], p[25], nullptr, xln2, BN, C);
  __bf16* Hb = (__bf16*)scratch.alloc((size_t)BN * 4 * C * 2);
  gemm(s, xln2, f1w, p[27], nullptr, nullptr, Hb, BN, 4 * C, C, C, 4 * C, 4 * C, 1);
  float* Xo = (float*)scratch.alloc((size_t)BN * C * 4);
  gemm(s, Hb, f2w, p[29], Xa, Xo, nullptr, BN, C, 4 * C, 4 * C, C, C, 0);

  // back to NCHW + input residual
  k_tok_to_img<<<gcfg((long)B * C * Np), 256, 0, s>>>(Xo, ximg, fimg_out, B, C, Np);
}

// 1x1 conv (as WMMA GEMM) + fxf average pool -> (B,Cout,H/f,W/f)
void conv1x1_pool(const float* fimg, const float* w, const float* bias,
                  int B, int Cin, int H, int W, int Cout, int f, float* outimg,
                  Arena scratch, hipStream_t s) {
  long Np = (long)H * W;
  float* tok = (float*)scratch.alloc((size_t)B * Np * Cin * 4);
  k_img_to_tok<<<gcfg((long)B * Cin * Np), 256, 0, s>>>(fimg, tok, B, Cin, (int)Np, Cin, 0);
  __bf16* tokb = (__bf16*)scratch.alloc((size_t)B * Np * Cin * 2);
  k_cvt_bf16<<<gcfg((long)B * Np * Cin), 256, 0, s>>>(tok, tokb, (long)B * Np * Cin);
  __bf16* wT = (__bf16*)scratch.alloc((size_t)Cin * Cout * 2);
  k_gather2d<<<gcfg((long)Cin * Cout), 256, 0, s>>>(w, nullptr, wT, Cin, Cout, 1, Cin);
  float* otok = (float*)scratch.alloc((size_t)B * Np * Cout * 4);
  gemm(s, tokb, wT, bias, nullptr, otok, nullptr, (long)B * Np, Cout, Cin, Cin, Cout,
       Cout, 0);
  k_avgpool_tok<<<gcfg((long)B * Cout * (H / f) * (W / f)), 256, 0, s>>>(
      otok, outimg, B, Cout, H, W, f);
}

void cross_attn(const float* xq_img, const float* kv_img, const float* f3img,
                const float* qw, const float* kvw, const float* pw, const float* pb,
                float* out_img, Arena scratch, hipStream_t s) {
  const int B = 8, C = 512, Np = 256;
  auto cvtw = [&](const float* w, long n) {
    __bf16* o = (__bf16*)scratch.alloc((size_t)n * 2);
    k_cvt_bf16<<<gcfg(n), 256, 0, s>>>(w, o, n);
    return o;
  };
  float* xt = (float*)scratch.alloc((size_t)B * Np * C * 4);
  k_img_to_tok<<<gcfg((long)B * C * Np), 256, 0, s>>>(xq_img, xt, B, C, Np, C, 0);
  __bf16* xtb = (__bf16*)scratch.alloc((size_t)B * Np * C * 2);
  k_cvt_bf16<<<gcfg((long)B * Np * C), 256, 0, s>>>(xt, xtb, (long)B * Np * C);
  float* ft = (float*)scratch.alloc((size_t)B * Np * C * 4);
  k_img_to_tok<<<gcfg((long)B * C * Np), 256, 0, s>>>(kv_img, ft, B, C, Np, C, 0);
  __bf16* ftb = (__bf16*)scratch.alloc((size_t)B * Np * C * 2);
  k_cvt_bf16<<<gcfg((long)B * Np * C), 256, 0, s>>>(ft, ftb, (long)B * Np * C);

  __bf16* qwb = cvtw(qw, (long)C * C);
  __bf16* kvwb = cvtw(kvw, (long)C * 2 * C);
  __bf16* pwb = cvtw(pw, (long)C * C);

  float* Qf = (float*)scratch.alloc((size_t)B * Np * C * 4);
  gemm(s, xtb, qwb, nullptr, nullptr, Qf, nullptr, (long)B * Np, C, C, C, C, C, 0);
  __bf16* Qb = (__bf16*)scratch.alloc((size_t)B * Np * C * 2);
  k_cvt_bf16<<<gcfg((long)B * Np * C), 256, 0, s>>>(Qf, Qb, (long)B * Np * C);
  float* KV = (float*)scratch.alloc((size_t)B * Np * 2 * C * 4);
  gemm(s, ftb, kvwb, nullptr, nullptr, KV, nullptr, (long)B * Np, 2 * C, C, C, 2 * C,
       2 * C, 0);

  __bf16* AO = (__bf16*)scratch.alloc((size_t)B * Np * C * 2);
  float* Sf = (float*)scratch.alloc((size_t)Np * Np * 4);
  __bf16* Pb2 = (__bf16*)scratch.alloc((size_t)Np * Np * 2);
  __bf16* KT = (__bf16*)scratch.alloc((size_t)C * Np * 2);
  __bf16* Vb = (__bf16*)scratch.alloc((size_t)Np * C * 2);
  float sc = 1.0f / sqrtf((float)C);
  for (int b = 0; b < B; ++b) {
    const float* kvb = KV + (long)b * Np * 2 * C;
    k_gather2d<<<gcfg((long)C * Np), 256, 0, s>>>(kvb, nullptr, KT, C, Np, 1, 2 * C);
    k_gather2d<<<gcfg((long)Np * C), 256, 0, s>>>(kvb + C, nullptr, Vb, Np, C, 2 * C, 1);
    gemm(s, Qb + (long)b * Np * C, KT, nullptr, nullptr, Sf, nullptr, Np, Np, C, C, Np,
         Np, 0);
    k_softmax<<<gcfg((long)Np), 256, 0, s>>>(Sf, Pb2, Np, Np, sc);
    gemm(s, Pb2, Vb, nullptr, nullptr, nullptr, AO + (long)b * Np * C, Np, C, Np, Np, C,
         C, 0);
  }
  float* Of = (float*)scratch.alloc((size_t)B * Np * C * 4);
  gemm(s, AO, pwb, pb, nullptr, Of, nullptr, (long)B * Np, C, C, C, C, C, 0);
  k_tok_to_img<<<gcfg((long)B * C * Np), 256, 0, s>>>(Of, f3img, out_img, B, C, Np);
}

}  // namespace

// ---------------------------------------------------------------------------
// Entry point
// ---------------------------------------------------------------------------
extern "C" void kernel_launch(void* const* d_in, const int* in_sizes, int n_in,
                              void* d_out, int out_size, void* d_ws, size_t ws_size,
                              hipStream_t stream) {
  (void)in_sizes; (void)n_in; (void)out_size; (void)ws_size;

  // Input indices per setup_inputs() flattening order.
  const float* X2 = (const float*)d_in[0];
  const float* X3 = (const float*)d_in[1];
  const float* X4 = (const float*)d_in[2];
  const int I_T1 = 3, I_T2 = 33, I_T3 = 63, I_CA = 93;
  const int I_C1W = 97, I_C1B = 98, I_C2W = 99, I_C2B = 100;
  const int I_FCW = 101, I_FCB = 102, I_FCMW = 103, I_FCMB = 104;

  const float* bp1[30]; const float* bp2[30]; const float* bp3[30];
  for (int i = 0; i < 30; ++i) {
    bp1[i] = (const float*)d_in[I_T1 + i];
    bp2[i] = (const float*)d_in[I_T2 + i];
    bp3[i] = (const float*)d_in[I_T3 + i];
  }

  // Persistent outputs live at the bottom of the workspace; per-phase scratch
  // above (each phase gets a fresh by-value arena).
  Arena persist{(char*)d_ws, 0};
  float* f1     = (float*)persist.alloc((size_t)8 * 128 * 64 * 64 * 4);
  float* f2     = (float*)persist.alloc((size_t)8 * 256 * 32 * 32 * 4);
  float* f3     = (float*)persist.alloc((size_t)8 * 512 * 16 * 16 * 4);
  float* f1p    = (float*)persist.alloc((size_t)8 * 512 * 16 * 16 * 4);
  float* f2p    = (float*)persist.alloc((size_t)8 * 512 * 16 * 16 * 4);
  float* fm_img = (float*)persist.alloc((size_t)8 * 512 * 16 * 16 * 4);
  float* fm     = (float*)persist.alloc((size_t)8 * 512 * 4);
  float* xg     = (float*)persist.alloc((size_t)8 * 512 * 4);
  Arena scratch{(char*)d_ws + (64ull << 20), 0};

  // Three transformer stages
  run_block(X2, bp1, 8, 128, 64, 64, 4, f1, scratch, stream);
  run_block(X3, bp2, 8, 256, 32, 32, 4, f2, scratch, stream);
  run_block(X4, bp3, 8, 512, 16, 16, 4, f3, scratch, stream);

  // 1x1 conv + pooling to 16x16
  conv1x1_pool(f1, (const float*)d_in[I_C1W], (const float*)d_in[I_C1B],
               8, 128, 64, 64, 512, 4, f1p, scratch, stream);
  conv1x1_pool(f2, (const float*)d_in[I_C2W], (const float*)d_in[I_C2B],
               8, 256, 32, 32, 512, 2, f2p, scratch, stream);

  // Cross attention fusing the three scales
  cross_attn(f1p, f2p, f3,
             (const float*)d_in[I_CA + 0], (const float*)d_in[I_CA + 1],
             (const float*)d_in[I_CA + 2], (const float*)d_in[I_CA + 3],
             fm_img, scratch, stream);

  // Heads
  k_mean_hw<<<gcfg(8 * 512), 256, 0, stream>>>(fm_img, fm, 8, 512, 256);
  k_mean_hw<<<gcfg(8 * 512), 256, 0, stream>>>(X4, xg, 8, 512, 256);
  k_final<<<gcfg(8 * 21), 256, 0, stream>>>(
      fm, xg, (const float*)d_in[I_FCW], (const float*)d_in[I_FCB],
      (const float*)d_in[I_FCMW], (const float*)d_in[I_FCMB],
      (float*)d_out, 8, 512, 21);
}
